// MultiheadAttention_37649683317083
// MI455X (gfx1250) — compile-verified, matrix operand order checked by reading, not run on hardware
//
#include <hip/hip_runtime.h>
#include <hip/hip_bf16.h>
#include <cstdint>

// ---------------------------------------------------------------------------
// MI455X (gfx1250) multi-head attention, f16 WMMA pipeline, f32 accumulate.
// GEMMs and attention stream operands via GLOBAL_LOAD_ASYNC_TO_LDS (ASYNCcnt)
// double-buffered against v_wmma_f32_16x16x32_f16.
// ---------------------------------------------------------------------------

typedef _Float16 half_t;
typedef __attribute__((ext_vector_type(2)))  _Float16 h2;
typedef __attribute__((ext_vector_type(4)))  _Float16 h4;
typedef __attribute__((ext_vector_type(8)))  _Float16 v8h;
typedef __attribute__((ext_vector_type(16))) _Float16 v16h;
typedef __attribute__((ext_vector_type(8)))  float    v8f;
typedef __attribute__((ext_vector_type(4)))  float    f4;

constexpr int Bc  = 2;
constexpr int Sq  = 2048;
constexpr int Dm  = 1024;
constexpr int Hh  = 16;
constexpr int DKc = 64;
constexpr int Mtot = Bc * Sq;         // 4096

constexpr int BM = 128, BN = 128, BKL = 64;      // GEMM tile; K-chunk = 64
constexpr int GSTR = 72;                         // GEMM LDS row stride (halves)
constexpr uint32_t GT_B = 128u * GSTR * 2;       // 18432 B per LDS buffer

// ---- WMMA helper ----------------------------------------------------------
__device__ __forceinline__ v8f wmma_f16(v16h a, v16h b, v8f c) {
  // D = A(16x32,f16) x B(32x16,f16) + C(16x16,f32)
  return __builtin_amdgcn_wmma_f32_16x16x32_f16(false, a, false, b,
                                                (short)0, c, false, false);
}

// K-index packing for 16-bit A/B fragments (ISA 7.12.2):
// lanes 0-15 : K = {0..7, 16..23};  lanes 16-31 : K = {8..15, 24..31}
__device__ __forceinline__ int frag_koff(int v, int laneHi) {
  return 2 * v + (v >= 4 ? 8 : 0) + (laneHi ? 8 : 0);
}

// Load one 16x32 fragment row (lane-local 16 halves) from a contiguous row.
__device__ __forceinline__ v16h load_frag_row(const half_t* row, int laneHi) {
  v16h f;
#pragma unroll
  for (int v = 0; v < 8; ++v) {
    h2 p = *reinterpret_cast<const h2*>(row + frag_koff(v, laneHi));
    f[2 * v]     = p.x;
    f[2 * v + 1] = p.y;
  }
  return f;
}

// 16 contiguous fp32 -> 16 contiguous f16 (LDS staging with conversion)
__device__ __forceinline__ void stage_f32(half_t* dst, const float* src) {
#pragma unroll
  for (int i = 0; i < 4; ++i) {
    f4 x = *reinterpret_cast<const f4*>(src + 4 * i);
    dst[4 * i + 0] = (half_t)x.x;
    dst[4 * i + 1] = (half_t)x.y;
    dst[4 * i + 2] = (half_t)x.z;
    dst[4 * i + 3] = (half_t)x.w;
  }
}

// ---------------------------------------------------------------------------
// Async DMA of one 128x64-half tile into LDS (row stride 72 halves = 144 B).
// 32 b128 transfers; each of 8 waves issues 4. GVS mode: SGPR64 base + v_off.
// ---------------------------------------------------------------------------
__device__ __forceinline__ void issue_gemm_tile_async(uint64_t gbase,
                                                      int rowStrideHalves,
                                                      int row0, int k0,
                                                      uint32_t ldsBase, int w,
                                                      int lane) {
#pragma unroll
  for (int i = 0; i < 4; ++i) {
    const int linear = (w * 4 + i) * 32 + lane;   // 8-half chunk id, 0..1023
    const int row  = linear >> 3;
    const int colh = (linear & 7) * 8;
    const uint32_t lds = ldsBase + (uint32_t)(row * 144 + colh * 2);
    const int voff = ((row0 + row) * rowStrideHalves + k0 + colh) * 2;
    asm volatile("global_load_async_to_lds_b128 %0, %1, %2"
                 :: "v"(lds), "v"(voff), "s"(gbase)
                 : "memory");
  }
}

// ---------------------------------------------------------------------------
// Kernel 0: fp32 -> f16 weight conversion (done once, not per M-tile).
// ---------------------------------------------------------------------------
__global__ void __launch_bounds__(256)
cvt_kernel(const float* __restrict__ src, half_t* __restrict__ dst, int n4) {
  const int i = blockIdx.x * blockDim.x + threadIdx.x;
  if (i < n4) {
    f4 x = reinterpret_cast<const f4*>(src)[i];
    h4 y;
    y.x = (half_t)x.x; y.y = (half_t)x.y; y.z = (half_t)x.z; y.w = (half_t)x.w;
    reinterpret_cast<h4*>(dst)[i] = y;
  }
}

// ---------------------------------------------------------------------------
// Kernel 1: projection GEMM. Y = X(4096x1024,f32) * W16^T(f16) + b, written
// f16 in head-major (B,H,S,DK) layout. A-side: fp32 convert-staging (must
// convert in flight). B-side: async DMA of pre-converted f16 weights.
// ---------------------------------------------------------------------------
__global__ void __launch_bounds__(256)
proj_kernel(const float* __restrict__ X, const half_t* __restrict__ W16,
            const float* __restrict__ bias, half_t* __restrict__ out) {
  __shared__ half_t As[2][BM][GSTR];
  __shared__ half_t Bs[2][BN][GSTR];

  const int t = threadIdx.x;
  const int lane = t & 31, lr = lane & 15, laneHi = lane >> 4;
  const int w = t >> 5, wm = w & 3, wn = w >> 2;   // 4x2 wave grid, 32x64/wave
  const int m0 = blockIdx.y * BM, n0 = blockIdx.x * BN;
  const int srow = t >> 1, skp = (t & 1) * 32;     // A staging: 32 floats/thread

  const uint64_t wbase = (uint64_t)(uintptr_t)W16;
  const uint32_t bsLds = (uint32_t)(uintptr_t)(&Bs[0][0][0]);

  v8f acc[2][4];
#pragma unroll
  for (int mt = 0; mt < 2; ++mt)
#pragma unroll
    for (int nt = 0; nt < 4; ++nt)
#pragma unroll
      for (int i = 0; i < 8; ++i) acc[mt][nt][i] = 0.0f;

  // prologue: chunk 0
  stage_f32(&As[0][srow][skp],      X + (size_t)(m0 + srow) * Dm + skp);
  stage_f32(&As[0][srow][skp + 16], X + (size_t)(m0 + srow) * Dm + skp + 16);
  issue_gemm_tile_async(wbase, Dm, n0, 0, bsLds, w, lane);

  const int NK = Dm / BKL;   // 16
  for (int kk = 0; kk < NK; ++kk) {
    const int buf = kk & 1;
    asm volatile("s_wait_asynccnt 0x0" ::: "memory");
    __syncthreads();
    if (kk + 1 < NK) {
      const int k0 = (kk + 1) * BKL;
      stage_f32(&As[buf ^ 1][srow][skp],
                X + (size_t)(m0 + srow) * Dm + k0 + skp);
      stage_f32(&As[buf ^ 1][srow][skp + 16],
                X + (size_t)(m0 + srow) * Dm + k0 + skp + 16);
      issue_gemm_tile_async(wbase, Dm, n0, k0,
                            bsLds + (uint32_t)(buf ^ 1) * GT_B, w, lane);
    }
#pragma unroll
    for (int s = 0; s < 2; ++s) {      // two K=32 sub-chunks
      v16h af[2], bf[4];
#pragma unroll
      for (int mt = 0; mt < 2; ++mt)
        af[mt] = load_frag_row(&As[buf][wm * 32 + mt * 16 + lr][s * 32], laneHi);
#pragma unroll
      for (int nt = 0; nt < 4; ++nt)
        bf[nt] = load_frag_row(&Bs[buf][wn * 64 + nt * 16 + lr][s * 32], laneHi);
#pragma unroll
      for (int mt = 0; mt < 2; ++mt)
#pragma unroll
        for (int nt = 0; nt < 4; ++nt)
          acc[mt][nt] = wmma_f16(af[mt], bf[nt], acc[mt][nt]);
    }
  }

  // epilogue: bias + f16 head-major scatter
#pragma unroll
  for (int mt = 0; mt < 2; ++mt) {
#pragma unroll
    for (int nt = 0; nt < 4; ++nt) {
      const int n = n0 + wn * 64 + nt * 16 + lr;
      const float bn = bias[n];
      const int hh = n >> 6, dk = n & 63;
      const int mbase = m0 + wm * 32 + mt * 16 + 8 * laneHi;
#pragma unroll
      for (int r = 0; r < 8; ++r) {
        const int m = mbase + r;
        const int bb = m >> 11, s = m & (Sq - 1);
        out[(((size_t)bb * Hh + hh) * Sq + s) * DKc + dk] =
            (half_t)(acc[mt][nt][r] + bn);
      }
    }
  }
}

// ---------------------------------------------------------------------------
// Async staging for attention: one 32x64-half (4 KB) tile -> LDS,
// 8x global_load_async_to_lds_b128 per tile, 2 issued by each of 4 waves.
// ---------------------------------------------------------------------------
__device__ __forceinline__ void issue_tile_async(uint64_t gbase, int halfOff,
                                                 uint32_t ldsBase, int w,
                                                 int lane) {
#pragma unroll
  for (int i = 0; i < 2; ++i) {
    const int linear = (w * 2 + i) * 32 + lane;        // 8-half chunk id, 0..255
    const int row  = linear >> 3;
    const int colh = (linear & 7) * 8;
    const uint32_t lds = ldsBase + (uint32_t)(row * 144 + colh * 2);
    const int voff = (halfOff + linear * 8) * 2;       // byte offset (GVS mode)
    asm volatile("global_load_async_to_lds_b128 %0, %1, %2"
                 :: "v"(lds), "v"(voff), "s"(gbase)
                 : "memory");
  }
}

// ---------------------------------------------------------------------------
// Kernel 2: flash attention per (b,h). S^T = K*Q^T so softmax is lane-local +
// one shfl_xor(16), and P^T maps directly onto the WMMA B-fragment packing.
// O^T = V^T * P^T with online softmax. K/V blocks stream into LDS via
// GLOBAL_LOAD_ASYNC_TO_LDS, double-buffered against WMMA (ASYNCcnt).
// Each wave owns 32 queries (2 tiles) so K/V fragments are reused 2x.
// ---------------------------------------------------------------------------
__global__ void __launch_bounds__(128)
attn_kernel(const half_t* __restrict__ qh, const half_t* __restrict__ kh,
            const half_t* __restrict__ vh, half_t* __restrict__ heads) {
  constexpr int BQ = 128, BKV = 32, NB = Sq / BKV;
  constexpr int STR = 72;                         // LDS row stride (halves)
  constexpr uint32_t TILE_B = (uint32_t)BKV * STR * 2;   // 4608 B per buffer
  __shared__ half_t Ks[2][BKV][STR];
  __shared__ half_t Vs[2][BKV][STR];

  const int t = threadIdx.x;
  const int lane = t & 31, lr = lane & 15, laneHi = lane >> 4;
  const int w = t >> 5;                           // 4 waves, 32 queries each
  const int bh = blockIdx.y;
  const size_t base = (size_t)bh * Sq * DKc;
  const int q0 = blockIdx.x * BQ + w * 32;

  const uint64_t kbase = (uint64_t)(uintptr_t)(kh + base);
  const uint64_t vbase = (uint64_t)(uintptr_t)(vh + base);
  const uint32_t ksLds = (uint32_t)(uintptr_t)(&Ks[0][0][0]);
  const uint32_t vsLds = (uint32_t)(uintptr_t)(&Vs[0][0][0]);

  // Q^T fragments (B-operand: lane = query column), kept in registers
  v16h qf[2][2];
#pragma unroll
  for (int qt = 0; qt < 2; ++qt) {
    const half_t* qrow = qh + base + (size_t)(q0 + qt * 16 + lr) * DKc;
#pragma unroll
    for (int c = 0; c < 2; ++c) qf[qt][c] = load_frag_row(qrow + c * 32, laneHi);
  }

  v8f o[2][4];
#pragma unroll
  for (int qt = 0; qt < 2; ++qt)
#pragma unroll
    for (int dt = 0; dt < 4; ++dt)
#pragma unroll
      for (int i = 0; i < 8; ++i) o[qt][dt][i] = 0.0f;
  float m_run[2] = {-1e30f, -1e30f};
  float l_run[2] = {0.0f, 0.0f};
  const float scale = 0.125f;   // 1/sqrt(64)

  // prologue: stream block 0 into buffer 0
  issue_tile_async(kbase, 0, ksLds, w, lane);
  issue_tile_async(vbase, 0, vsLds, w, lane);

  for (int kb = 0; kb < NB; ++kb) {
    const int buf = kb & 1;
    asm volatile("s_wait_asynccnt 0x0" ::: "memory");  // my tile portion landed
    __syncthreads();                                   // everyone's landed
    if (kb + 1 < NB) {                                 // stream next block now
      issue_tile_async(kbase, (kb + 1) * (BKV * DKc),
                       ksLds + (uint32_t)(buf ^ 1) * TILE_B, w, lane);
      issue_tile_async(vbase, (kb + 1) * (BKV * DKc),
                       vsLds + (uint32_t)(buf ^ 1) * TILE_B, w, lane);
    }

    const half_t* ksB = &Ks[buf][0][0];
    const half_t* vsB = &Vs[buf][0][0];

    // K fragments (A-operand: lane = key row), reused by both query tiles
    v16h kf[2][2];
#pragma unroll
    for (int kt = 0; kt < 2; ++kt)
#pragma unroll
      for (int c = 0; c < 2; ++c)
        kf[kt][c] = load_frag_row(ksB + (kt * 16 + lr) * STR + c * 32, laneHi);

    v16h pf[2];   // P^T B-fragments for both query tiles
#pragma unroll
    for (int qt = 0; qt < 2; ++qt) {
      v8f st[2];
#pragma unroll
      for (int kt = 0; kt < 2; ++kt) {
        v8f s;
#pragma unroll
        for (int i = 0; i < 8; ++i) s[i] = 0.0f;
#pragma unroll
        for (int c = 0; c < 2; ++c) s = wmma_f16(kf[kt][c], qf[qt][c], s);
        st[kt] = s;
      }
      // online softmax along keys (per query = lane pair {q, q^16})
      float mloc = -1e30f;
#pragma unroll
      for (int kt = 0; kt < 2; ++kt)
#pragma unroll
        for (int i = 0; i < 8; ++i) {
          const float vsc = st[kt][i] * scale;
          st[kt][i] = vsc;
          mloc = fmaxf(mloc, vsc);
        }
      mloc = fmaxf(mloc, __shfl_xor(mloc, 16));
      const float mnew = fmaxf(m_run[qt], mloc);
      const float corr = __expf(m_run[qt] - mnew);
      float lloc = 0.0f;
#pragma unroll
      for (int kt = 0; kt < 2; ++kt)
#pragma unroll
        for (int i = 0; i < 8; ++i) {
          const float e = __expf(st[kt][i] - mnew);
          lloc += e;
          pf[qt][kt * 8 + i] = (half_t)e;
        }
      lloc += __shfl_xor(lloc, 16);
      l_run[qt] = l_run[qt] * corr + lloc;
      m_run[qt] = mnew;
#pragma unroll
      for (int dt = 0; dt < 4; ++dt)
#pragma unroll
        for (int i = 0; i < 8; ++i) o[qt][dt][i] *= corr;
    }

    // O^T += V^T x P^T : V^T fragment gathered from untransposed V in LDS
#pragma unroll
    for (int dt = 0; dt < 4; ++dt) {
      const int d = dt * 16 + lr;
      v16h vf;
#pragma unroll
      for (int v = 0; v < 8; ++v) {
        const int k = frag_koff(v, laneHi);
        vf[2 * v]     = vsB[k * STR + d];
        vf[2 * v + 1] = vsB[(k + 1) * STR + d];
      }
#pragma unroll
      for (int qt = 0; qt < 2; ++qt)
        o[qt][dt] = wmma_f16(vf, pf[qt], o[qt][dt]);
    }
  }

  // normalize and store heads (B,H,S,DK) f16; 8 contiguous d per lane -> b128
#pragma unroll
  for (int qt = 0; qt < 2; ++qt) {
    const float inv = 1.0f / l_run[qt];
#pragma unroll
    for (int dt = 0; dt < 4; ++dt) {
      v8h ov;
#pragma unroll
      for (int r = 0; r < 8; ++r) ov[r] = (half_t)(o[qt][dt][r] * inv);
      const size_t idx =
          base + (size_t)(q0 + qt * 16 + lr) * DKc + dt * 16 + 8 * laneHi;
      *reinterpret_cast<v8h*>(heads + idx) = ov;
    }
  }
}

// ---------------------------------------------------------------------------
// Kernel 3: output projection. A = heads f16 flat (4096x1024) (the reference's
// raw reshape of head-major == our flat buffer), W = Wo pre-converted f16,
// out f32. Both operands stream via async DMA; inner loop is pure WMMA.
// ---------------------------------------------------------------------------
__global__ void __launch_bounds__(256)
outproj_kernel(const half_t* __restrict__ A, const half_t* __restrict__ W16,
               const float* __restrict__ bias, float* __restrict__ out) {
  __shared__ half_t As[2][BM][GSTR];
  __shared__ half_t Bs[2][BN][GSTR];

  const int t = threadIdx.x;
  const int lane = t & 31, lr = lane & 15, laneHi = lane >> 4;
  const int w = t >> 5, wm = w & 3, wn = w >> 2;
  const int m0 = blockIdx.y * BM, n0 = blockIdx.x * BN;

  const uint64_t abase = (uint64_t)(uintptr_t)A;
  const uint64_t wbase = (uint64_t)(uintptr_t)W16;
  const uint32_t asLds = (uint32_t)(uintptr_t)(&As[0][0][0]);
  const uint32_t bsLds = (uint32_t)(uintptr_t)(&Bs[0][0][0]);

  v8f acc[2][4];
#pragma unroll
  for (int mt = 0; mt < 2; ++mt)
#pragma unroll
    for (int nt = 0; nt < 4; ++nt)
#pragma unroll
      for (int i = 0; i < 8; ++i) acc[mt][nt][i] = 0.0f;

  issue_gemm_tile_async(abase, Dm, m0, 0, asLds, w, lane);
  issue_gemm_tile_async(wbase, Dm, n0, 0, bsLds, w, lane);

  const int NK = Dm / BKL;   // 16
  for (int kk = 0; kk < NK; ++kk) {
    const int buf = kk & 1;
    asm volatile("s_wait_asynccnt 0x0" ::: "memory");
    __syncthreads();
    if (kk + 1 < NK) {
      const int k0 = (kk + 1) * BKL;
      issue_gemm_tile_async(abase, Dm, m0, k0,
                            asLds + (uint32_t)(buf ^ 1) * GT_B, w, lane);
      issue_gemm_tile_async(wbase, Dm, n0, k0,
                            bsLds + (uint32_t)(buf ^ 1) * GT_B, w, lane);
    }
#pragma unroll
    for (int s = 0; s < 2; ++s) {
      v16h af[2], bf[4];
#pragma unroll
      for (int mt = 0; mt < 2; ++mt)
        af[mt] = load_frag_row(&As[buf][wm * 32 + mt * 16 + lr][s * 32], laneHi);
#pragma unroll
      for (int nt = 0; nt < 4; ++nt)
        bf[nt] = load_frag_row(&Bs[buf][wn * 64 + nt * 16 + lr][s * 32], laneHi);
#pragma unroll
      for (int mt = 0; mt < 2; ++mt)
#pragma unroll
        for (int nt = 0; nt < 4; ++nt)
          acc[mt][nt] = wmma_f16(af[mt], bf[nt], acc[mt][nt]);
    }
  }

#pragma unroll
  for (int mt = 0; mt < 2; ++mt) {
#pragma unroll
    for (int nt = 0; nt < 4; ++nt) {
      const int n = n0 + wn * 64 + nt * 16 + lr;
      const float bn = bias[n];
      const int mbase = m0 + wm * 32 + mt * 16 + 8 * laneHi;
#pragma unroll
      for (int r = 0; r < 8; ++r)
        out[(size_t)(mbase + r) * Dm + n] = acc[mt][nt][r] + bn;
    }
  }
}

// ---------------------------------------------------------------------------
extern "C" void kernel_launch(void* const* d_in, const int* in_sizes, int n_in,
                              void* d_out, int out_size, void* d_ws,
                              size_t ws_size, hipStream_t stream) {
  const float* query = (const float*)d_in[0];
  const float* key   = (const float*)d_in[1];
  const float* value = (const float*)d_in[2];
  const float* Wq = (const float*)d_in[3];
  const float* bq = (const float*)d_in[4];
  const float* Wk = (const float*)d_in[5];
  const float* bk = (const float*)d_in[6];
  const float* Wv = (const float*)d_in[7];
  const float* bv = (const float*)d_in[8];
  const float* Wo = (const float*)d_in[9];
  const float* bo = (const float*)d_in[10];
  float* out = (float*)d_out;

  half_t* ws = (half_t*)d_ws;
  const size_t region = (size_t)Bc * Hh * Sq * DKc;   // 4,194,304 halves (8 MB)
  const size_t wreg = (size_t)Dm * Dm;                // 1,048,576 halves (2 MB)
  half_t* qhb  = ws;
  half_t* khb  = ws + region;
  half_t* vhb  = ws + 2 * region;
  half_t* hdb  = ws + 3 * region;
  half_t* wq16 = ws + 4 * region;
  half_t* wk16 = wq16 + wreg;
  half_t* wv16 = wq16 + 2 * wreg;
  half_t* wo16 = wq16 + 3 * wreg;

  // convert weights fp32 -> f16 once
  const int n4 = (int)(wreg / 4);   // 262144
  cvt_kernel<<<n4 / 256, 256, 0, stream>>>(Wq, wq16, n4);
  cvt_kernel<<<n4 / 256, 256, 0, stream>>>(Wk, wk16, n4);
  cvt_kernel<<<n4 / 256, 256, 0, stream>>>(Wv, wv16, n4);
  cvt_kernel<<<n4 / 256, 256, 0, stream>>>(Wo, wo16, n4);

  dim3 gridG(Dm / BN, Mtot / BM);   // (8, 32)
  proj_kernel<<<gridG, 256, 0, stream>>>(query, wq16, bq, qhb);
  proj_kernel<<<gridG, 256, 0, stream>>>(key,   wk16, bk, khb);
  proj_kernel<<<gridG, 256, 0, stream>>>(value, wv16, bv, vhb);

  dim3 gridA(Sq / 128, Bc * Hh);    // (16, 32)
  attn_kernel<<<gridA, 128, 0, stream>>>(qhb, khb, vhb, hdb);

  outproj_kernel<<<gridG, 256, 0, stream>>>(hdb, wo16, bo, out);
}
